// Seq2Seq_75668733821270
// MI455X (gfx1250) — compile-verified, matrix operand order checked by reading, not run on hardware
//
#include <hip/hip_runtime.h>

typedef __attribute__((ext_vector_type(16))) __bf16 v16bf;
typedef __attribute__((ext_vector_type(8)))  __bf16 v8bf;
typedef __attribute__((ext_vector_type(8)))  float  v8f;

#define SCAN_BLOCKS 8
#define SCAN_THREADS 256

// ---------- helpers ----------
static __device__ __forceinline__ __bf16 tobf(float f) {
  unsigned u = __float_as_uint(f);
  unsigned r = u + 0x7FFFu + ((u >> 16) & 1u);   // round-to-nearest-even
  unsigned short hs = (unsigned short)(r >> 16);
  __bf16 o; __builtin_memcpy(&o, &hs, 2); return o;
}
static __device__ __forceinline__ float sigf(float x) { return 1.0f / (1.0f + __expf(-x)); }
static __device__ __forceinline__ float tanh_f(float x) {
  x = fminf(fmaxf(x, -15.0f), 15.0f);
  float e = __expf(2.0f * x);
  return (e - 1.0f) / (e + 1.0f);
}

// 16-bit A/B fragment, row-major [rows x K], ld = row stride (elements).
// Lane l: row = r0 + (l&15); lanes 0-15 hold K in {k0..k0+7, k0+16..k0+23},
// lanes 16-31 hold K in {k0+8..k0+15, k0+24..k0+31}  (ISA 7.12.2).
static __device__ __forceinline__ const __bf16* frag_addr(const __bf16* P, int ld,
                                                          int r0, int k0, int lane) {
  return P + (size_t)(r0 + (lane & 15)) * ld + k0 + ((lane >> 4) & 1) * 8;
}
static __device__ __forceinline__ v16bf frag_rm(const __bf16* __restrict__ P, int ld,
                                                int r0, int k0, int lane) {
  const __bf16* p = frag_addr(P, ld, r0, k0, lane);
  v8bf lo = *(const v8bf*)(p);
  v8bf hi = *(const v8bf*)(p + 16);
  v16bf f;
#pragma unroll
  for (int i = 0; i < 8; i++) { f[i] = lo[i]; f[i + 8] = hi[i]; }
  return f;
}
// B fragment from row-major [K x N] (non-transposed B): strided gather.
static __device__ __forceinline__ v16bf frag_cm(const __bf16* __restrict__ P, int ld,
                                                int k0, int n0, int lane) {
  int n = lane & 15, khi = (lane >> 4) & 1;
  v16bf f;
#pragma unroll
  for (int i = 0; i < 8; i++) {
    f[i]     = P[(size_t)(k0 + khi * 8 + i) * ld + n0 + n];
    f[i + 8] = P[(size_t)(k0 + 16 + khi * 8 + i) * ld + n0 + n];
  }
  return f;
}
static __device__ __forceinline__ v8f wmma_bf16(v16bf a, v16bf b, v8f c) {
  return __builtin_amdgcn_wmma_f32_16x16x32_bf16(false, a, false, b, (short)0, c,
                                                 false, false);
}

// ---------- generic WMMA GEMM: C[M,N] = A[M,K](bf16) * B(bf16) + bias ----------
// BT=1: B is [N,K] row-major (X @ W^T). BT=0: B is [K,N] row-major.
// Wave computes a 32x64 tile. Explicitly double-buffered: fragments for k+32
// are loaded into fresh values before the 8 WMMAs for k issue, forcing distinct
// registers so loads pipeline under WMMAs. __launch_bounds__(256) raises the
// per-wave VGPR cap (we run 8 waves/block) so nothing spills to scratch.
template <int BT>
__global__ void __launch_bounds__(256)
wmma_gemm_kernel(const __bf16* __restrict__ A, const __bf16* __restrict__ Bm,
                 const float* __restrict__ bias, float* __restrict__ C,
                 int M, int N, int K, int lda, int ldb, int ldc,
                 long sA, long sB, long sC) {
  A  += (size_t)blockIdx.z * sA;
  Bm += (size_t)blockIdx.z * sB;
  C  += (size_t)blockIdx.z * sC;
  int lane = threadIdx.x & 31;
  int w = blockIdx.x * (blockDim.x >> 5) + (threadIdx.x >> 5);
  int n4tiles = N >> 6;
  int mtile = w / n4tiles, n4 = w - mtile * n4tiles;
  if (mtile >= (M >> 5)) return;
  int m0 = mtile << 5, n0 = n4 << 6;
  int nlo = lane & 15, mhi = (lane >> 4) * 8;

  v8f acc[2][4];
#pragma unroll
  for (int j = 0; j < 4; j++) {
    float b0 = bias ? bias[n0 + j * 16 + nlo] : 0.0f;
#pragma unroll
    for (int r = 0; r < 8; r++) { acc[0][j][r] = b0; acc[1][j][r] = b0; }
  }

  v16bf a0 = frag_rm(A, lda, m0, 0, lane);
  v16bf a1 = frag_rm(A, lda, m0 + 16, 0, lane);
  v16bf b0 = BT ? frag_rm(Bm, ldb, n0 +  0, 0, lane) : frag_cm(Bm, ldb, 0, n0 +  0, lane);
  v16bf b1 = BT ? frag_rm(Bm, ldb, n0 + 16, 0, lane) : frag_cm(Bm, ldb, 0, n0 + 16, lane);
  v16bf b2 = BT ? frag_rm(Bm, ldb, n0 + 32, 0, lane) : frag_cm(Bm, ldb, 0, n0 + 32, lane);
  v16bf b3 = BT ? frag_rm(Bm, ldb, n0 + 48, 0, lane) : frag_cm(Bm, ldb, 0, n0 + 48, lane);

  for (int k0 = 0; k0 < K - 32; k0 += 32) {
    int kn = k0 + 32;
    if (kn + 256 < K) {  // warm L2/WGP$ for the streaming operands
      __builtin_prefetch(frag_addr(A, lda, m0, kn + 256, lane), 0, 1);
      if (BT) __builtin_prefetch(frag_addr(Bm, ldb, n0, kn + 256, lane), 0, 1);
    }
    v16bf na0 = frag_rm(A, lda, m0, kn, lane);
    v16bf na1 = frag_rm(A, lda, m0 + 16, kn, lane);
    v16bf nb0 = BT ? frag_rm(Bm, ldb, n0 +  0, kn, lane) : frag_cm(Bm, ldb, kn, n0 +  0, lane);
    v16bf nb1 = BT ? frag_rm(Bm, ldb, n0 + 16, kn, lane) : frag_cm(Bm, ldb, kn, n0 + 16, lane);
    v16bf nb2 = BT ? frag_rm(Bm, ldb, n0 + 32, kn, lane) : frag_cm(Bm, ldb, kn, n0 + 32, lane);
    v16bf nb3 = BT ? frag_rm(Bm, ldb, n0 + 48, kn, lane) : frag_cm(Bm, ldb, kn, n0 + 48, lane);
    acc[0][0] = wmma_bf16(a0, b0, acc[0][0]);
    acc[1][0] = wmma_bf16(a1, b0, acc[1][0]);
    acc[0][1] = wmma_bf16(a0, b1, acc[0][1]);
    acc[1][1] = wmma_bf16(a1, b1, acc[1][1]);
    acc[0][2] = wmma_bf16(a0, b2, acc[0][2]);
    acc[1][2] = wmma_bf16(a1, b2, acc[1][2]);
    acc[0][3] = wmma_bf16(a0, b3, acc[0][3]);
    acc[1][3] = wmma_bf16(a1, b3, acc[1][3]);
    a0 = na0; a1 = na1; b0 = nb0; b1 = nb1; b2 = nb2; b3 = nb3;
  }
  acc[0][0] = wmma_bf16(a0, b0, acc[0][0]);
  acc[1][0] = wmma_bf16(a1, b0, acc[1][0]);
  acc[0][1] = wmma_bf16(a0, b1, acc[0][1]);
  acc[1][1] = wmma_bf16(a1, b1, acc[1][1]);
  acc[0][2] = wmma_bf16(a0, b2, acc[0][2]);
  acc[1][2] = wmma_bf16(a1, b2, acc[1][2]);
  acc[0][3] = wmma_bf16(a0, b3, acc[0][3]);
  acc[1][3] = wmma_bf16(a1, b3, acc[1][3]);

#pragma unroll
  for (int h = 0; h < 2; h++)
#pragma unroll
    for (int j = 0; j < 4; j++)
#pragma unroll
      for (int r = 0; r < 8; r++)
        C[(size_t)(m0 + h * 16 + mhi + r) * ldc + n0 + j * 16 + nlo] = acc[h][j][r];
}

// ---------- device-wide barrier (grid fully resident: 8 blocks) ----------
static __device__ __forceinline__ void gbarrier(int* bar) {
  __syncthreads();
  __threadfence();   // every wave writes back its dirty lines before signaling
  if (threadIdx.x == 0) {
    int g = __hip_atomic_load(bar + 1, __ATOMIC_RELAXED, __HIP_MEMORY_SCOPE_AGENT);
    if (__hip_atomic_fetch_add(bar, 1, __ATOMIC_ACQ_REL, __HIP_MEMORY_SCOPE_AGENT) ==
        SCAN_BLOCKS - 1) {
      __hip_atomic_store(bar, 0, __ATOMIC_RELAXED, __HIP_MEMORY_SCOPE_AGENT);
      __hip_atomic_fetch_add(bar + 1, 1, __ATOMIC_RELEASE, __HIP_MEMORY_SCOPE_AGENT);
    } else {
      while (__hip_atomic_load(bar + 1, __ATOMIC_ACQUIRE, __HIP_MEMORY_SCOPE_AGENT) == g)
        __builtin_amdgcn_s_sleep(2);
    }
  }
  __syncthreads();
}

// ---------- persistent LSTM scan, ONE barrier per step ----------
// 64 waves; wave w owns H-columns [16w,16w+16) of ALL FOUR gates:
// tiles at N = n0, 1024+n0, 2048+n0, 3072+n0 (K=1024 -> 4x32 WMMA/wave/step).
// C/D layout puts i/f/g/o for element (m,n) in the same lane & register index,
// so gates + state update run entirely in registers: no z buffer, no 2nd barrier.
// K-loop explicitly double-buffered like the GEMM.
__global__ void __launch_bounds__(SCAN_THREADS)
lstm_scan_kernel(const float* __restrict__ xW, const __bf16* __restrict__ Whh,
                 int steps, float* __restrict__ h, float* __restrict__ c,
                 __bf16* __restrict__ hb,
                 float* __restrict__ out, int out_ld, int* __restrict__ bar) {
  const int lane = threadIdx.x & 31;
  const int w = blockIdx.x * (SCAN_THREADS / 32) + (threadIdx.x >> 5); // 0..63
  const int n0 = w << 4;                                               // H column base
  const int nlo = lane & 15, mhi = (lane >> 4) * 8;
  for (int t = 0; t < steps; ++t) {
    v8f acc[4];
#pragma unroll
    for (int j = 0; j < 4; ++j)
#pragma unroll
      for (int r = 0; r < 8; ++r)
        acc[j][r] = xW[(size_t)((mhi + r) * steps + t) * 4096 + j * 1024 + n0 + nlo];

    v16bf a  = frag_rm(hb, 1024, 0, 0, lane);    // h: 16x1024 (batch = one M-tile)
    v16bf b0 = frag_rm(Whh, 1024,        n0, 0, lane);
    v16bf b1 = frag_rm(Whh, 1024, 1024 + n0, 0, lane);
    v16bf b2 = frag_rm(Whh, 1024, 2048 + n0, 0, lane);
    v16bf b3 = frag_rm(Whh, 1024, 3072 + n0, 0, lane);
    for (int k0 = 0; k0 < 1024 - 32; k0 += 32) {
      int kn = k0 + 32;
      v16bf na  = frag_rm(hb, 1024, 0, kn, lane);
      v16bf nb0 = frag_rm(Whh, 1024,        n0, kn, lane);
      v16bf nb1 = frag_rm(Whh, 1024, 1024 + n0, kn, lane);
      v16bf nb2 = frag_rm(Whh, 1024, 2048 + n0, kn, lane);
      v16bf nb3 = frag_rm(Whh, 1024, 3072 + n0, kn, lane);
      acc[0] = wmma_bf16(a, b0, acc[0]);
      acc[1] = wmma_bf16(a, b1, acc[1]);
      acc[2] = wmma_bf16(a, b2, acc[2]);
      acc[3] = wmma_bf16(a, b3, acc[3]);
      a = na; b0 = nb0; b1 = nb1; b2 = nb2; b3 = nb3;
    }
    acc[0] = wmma_bf16(a, b0, acc[0]);
    acc[1] = wmma_bf16(a, b1, acc[1]);
    acc[2] = wmma_bf16(a, b2, acc[2]);
    acc[3] = wmma_bf16(a, b3, acc[3]);

    // gates + state update in registers (i,f,g,o gate order as in PyTorch)
#pragma unroll
    for (int r = 0; r < 8; ++r) {
      int m = mhi + r;
      size_t idx = (size_t)m * 1024 + n0 + nlo;
      float iv = sigf(acc[0][r]);
      float fv = sigf(acc[1][r]);
      float gv = tanh_f(acc[2][r]);
      float ov = sigf(acc[3][r]);
      float cn = fv * c[idx] + iv * gv;
      float hn = ov * tanh_f(cn);
      c[idx] = cn; h[idx] = hn; hb[idx] = tobf(hn);
      out[(size_t)(m * steps + t) * out_ld + n0 + nlo] = hn;
    }
    gbarrier(bar);   // single barrier: h/c/hb visible before next step's GEMM
  }
}

// ---------- small elementwise / setup kernels ----------
__global__ void init_state_kernel(float* h, float* c, __bf16* hb, int* bar, int n) {
  int i = blockIdx.x * blockDim.x + threadIdx.x;
  if (i < n) { h[i] = 0.0f; c[i] = 0.0f; hb[i] = tobf(0.0f); }
  if (i < 8) bar[i] = 0;
}
__global__ void bias_sum_kernel(const float* __restrict__ a, const float* __restrict__ b,
                                float* __restrict__ o, int n) {
  int i = blockIdx.x * blockDim.x + threadIdx.x;
  if (i < n) o[i] = a[i] + b[i];
}
__global__ void cvt_kernel(const float* __restrict__ src, long sld,
                           __bf16* __restrict__ dst, long dld, long rows, long cols) {
  long i = (long)blockIdx.x * 256 + threadIdx.x;
  if (i >= rows * cols) return;
  long r = i / cols, cc = i - r * cols;
  dst[r * dld + cc] = tobf(src[r * sld + cc]);
}
__global__ void embed_bf16_kernel(const int* __restrict__ ids, const float* __restrict__ emb,
                                  __bf16* __restrict__ dst, long n, int E) {
  long i = (long)blockIdx.x * 256 + threadIdx.x;
  if (i >= n) return;
  long row = i / E; int e = (int)(i - row * E);
  dst[i] = tobf(emb[(size_t)ids[row] * E + e]);
}
__global__ void embed_dec_kernel(const int* __restrict__ ids, const float* __restrict__ emb,
                                 float* __restrict__ attn, long n) {
  long i = (long)blockIdx.x * 256 + threadIdx.x;
  if (i >= n) return;
  long bt = i >> 9; int e = (int)(i & 511);
  attn[bt * 3072 + 2048 + e] = emb[(size_t)ids[bt] * 512 + e];
}
__global__ void context_kernel(const float* __restrict__ h, const float* __restrict__ c,
                               float* __restrict__ attn, long n) {
  long i = (long)blockIdx.x * 256 + threadIdx.x;
  if (i >= n) return;
  long bt = i >> 10; int nn = (int)(i & 1023); int b = (int)(bt >> 7);
  attn[bt * 3072 + 1024 + nn] = h[b * 1024 + nn] + c[b * 1024 + nn];
}
__global__ void __launch_bounds__(256)
softmax_kernel(const float* __restrict__ x, __bf16* __restrict__ o, int cols) {
  long row = blockIdx.x;
  const float* xr = x + (size_t)row * cols;
  __shared__ float red[256];
  float m = -3.4e38f;
  for (int i = threadIdx.x; i < cols; i += 256) m = fmaxf(m, xr[i]);
  red[threadIdx.x] = m; __syncthreads();
  for (int s = 128; s > 0; s >>= 1) {
    if (threadIdx.x < s) red[threadIdx.x] = fmaxf(red[threadIdx.x], red[threadIdx.x + s]);
    __syncthreads();
  }
  m = red[0]; __syncthreads();
  float sum = 0.0f;
  for (int i = threadIdx.x; i < cols; i += 256) sum += __expf(xr[i] - m);
  red[threadIdx.x] = sum; __syncthreads();
  for (int s = 128; s > 0; s >>= 1) {
    if (threadIdx.x < s) red[threadIdx.x] += red[threadIdx.x + s];
    __syncthreads();
  }
  float inv = 1.0f / red[0];
  for (int i = threadIdx.x; i < cols; i += 256)
    o[(size_t)row * cols + i] = tobf(__expf(xr[i] - m) * inv);
}
__global__ void gate_kernel(const float* __restrict__ attn, const float* __restrict__ htb,
                            const float* __restrict__ ail, __bf16* __restrict__ og, long n) {
  long i = (long)blockIdx.x * 256 + threadIdx.x;
  if (i >= n) return;
  long r = i >> 10; int nn = (int)(i & 1023);
  float d = attn[r * 3072 + nn];        // dec_out
  og[i] = tobf(d + htb[i] * sigf(ail[i]));
}
__global__ void targets_kernel(float* __restrict__ t, long n) {
  long i = (long)blockIdx.x * 256 + threadIdx.x;
  if (i >= n) return;
  t[i] = (float)(4 * (i & 127));        // (512/128)*t
}

// ---------- host ----------
extern "C" void kernel_launch(void* const* d_in, const int* in_sizes, int n_in,
                              void* d_out, int out_size, void* d_ws, size_t ws_size,
                              hipStream_t stream) {
  (void)in_sizes; (void)n_in; (void)out_size; (void)ws_size;
  const int*   x_enc = (const int*)d_in[0];
  const int*   x_dec = (const int*)d_in[1];
  const float* emb   = (const float*)d_in[4];
  const float* Wih_e = (const float*)d_in[5];
  const float* Whh_e = (const float*)d_in[6];
  const float* bih_e = (const float*)d_in[7];
  const float* bhh_e = (const float*)d_in[8];
  const float* Wih_d = (const float*)d_in[9];
  const float* Whh_d = (const float*)d_in[10];
  const float* bih_d = (const float*)d_in[11];
  const float* bhh_d = (const float*)d_in[12];
  const float* Wa    = (const float*)d_in[13];
  const float* ba    = (const float*)d_in[14];
  const float* Wal   = (const float*)d_in[15];
  const float* bal   = (const float*)d_in[16];
  const float* Wo    = (const float*)d_in[17];
  const float* bo    = (const float*)d_in[18];

  float* out_v   = (float*)d_out;                    // [2048, 32000]
  float* attn_in = out_v + (size_t)2048 * 32000;     // [2048, 3072] (dec_out|ctx|dec_x)
  float* tgt     = attn_in + (size_t)2048 * 3072;    // [2048]

  char* base = (char*)d_ws; size_t off = 0;
  auto alloc = [&](size_t bytes) -> void* {
    void* p = base + off; off += (bytes + 255) & ~(size_t)255; return p; };

  // persistent
  __bf16* bWih_e = (__bf16*)alloc((size_t)4096 * 512 * 2);
  __bf16* bWhh_e = (__bf16*)alloc((size_t)4096 * 1024 * 2);
  __bf16* bWih_d = (__bf16*)alloc((size_t)4096 * 1536 * 2);
  __bf16* bWhh_d = (__bf16*)alloc((size_t)4096 * 1024 * 2);
  __bf16* bWa    = (__bf16*)alloc((size_t)512 * 3072 * 2);
  __bf16* bWal   = (__bf16*)alloc((size_t)1024 * 2048 * 2);
  __bf16* bWo    = (__bf16*)alloc((size_t)32000 * 1024 * 2);
  float*  bsum_e = (float*)alloc(4096 * 4);
  float*  bsum_d = (float*)alloc(4096 * 4);
  float*  hS     = (float*)alloc(16 * 1024 * 4);
  float*  cS     = (float*)alloc(16 * 1024 * 4);
  __bf16* hB     = (__bf16*)alloc(16 * 1024 * 2);
  int*    bar    = (int*)alloc(256);
  float*  enc_out   = (float*)alloc((size_t)8192 * 1024 * 4);
  __bf16* enc_out_b = (__bf16*)alloc((size_t)8192 * 1024 * 2);
  size_t arena = off;
  // phase A (encoder)
  __bf16* enc_x = (__bf16*)alloc((size_t)8192 * 512 * 2);
  float*  xW_e  = (float*)alloc((size_t)8192 * 4096 * 4);
  // phase B (decoder) overlays phase A
  off = arena;
  __bf16* lstm_in = (__bf16*)alloc((size_t)2048 * 1536 * 2);
  float*  xW_d    = (float*)alloc((size_t)2048 * 4096 * 4);
  __bf16* attn_b  = (__bf16*)alloc((size_t)2048 * 3072 * 2);
  float*  logits  = (float*)alloc((size_t)2048 * 512 * 4);
  __bf16* softw   = (__bf16*)alloc((size_t)2048 * 512 * 2);
  float*  htbar   = (float*)alloc((size_t)2048 * 1024 * 4);
  __bf16* catg    = (__bf16*)alloc((size_t)2048 * 2048 * 2);
  float*  ailog   = (float*)alloc((size_t)2048 * 1024 * 4);
  __bf16* outg    = (__bf16*)alloc((size_t)2048 * 1024 * 2);

  auto g1 = [](long n) { return dim3((unsigned)((n + 255) / 256)); };

  // setup
  init_state_kernel<<<g1(16 * 1024), 256, 0, stream>>>(hS, cS, hB, bar, 16 * 1024);
  bias_sum_kernel<<<g1(4096), 256, 0, stream>>>(bih_e, bhh_e, bsum_e, 4096);
  bias_sum_kernel<<<g1(4096), 256, 0, stream>>>(bih_d, bhh_d, bsum_d, 4096);
  cvt_kernel<<<g1((long)4096 * 512), 256, 0, stream>>>(Wih_e, 512, bWih_e, 512, 4096, 512);
  cvt_kernel<<<g1((long)4096 * 1024), 256, 0, stream>>>(Whh_e, 1024, bWhh_e, 1024, 4096, 1024);
  cvt_kernel<<<g1((long)4096 * 1536), 256, 0, stream>>>(Wih_d, 1536, bWih_d, 1536, 4096, 1536);
  cvt_kernel<<<g1((long)4096 * 1024), 256, 0, stream>>>(Whh_d, 1024, bWhh_d, 1024, 4096, 1024);
  cvt_kernel<<<g1((long)512 * 3072), 256, 0, stream>>>(Wa, 3072, bWa, 3072, 512, 3072);
  cvt_kernel<<<g1((long)1024 * 2048), 256, 0, stream>>>(Wal, 2048, bWal, 2048, 1024, 2048);
  cvt_kernel<<<g1((long)32000 * 1024), 256, 0, stream>>>(Wo, 1024, bWo, 1024, 32000, 1024);

  // encoder
  embed_bf16_kernel<<<g1((long)8192 * 512), 256, 0, stream>>>(x_enc, emb, enc_x,
                                                              (long)8192 * 512, 512);
  { int waves = (8192 / 32) * (4096 / 64);
    wmma_gemm_kernel<1><<<dim3((waves + 7) / 8), 256, 0, stream>>>(
        enc_x, bWih_e, bsum_e, xW_e, 8192, 4096, 512, 512, 512, 4096, 0, 0, 0); }
  lstm_scan_kernel<<<dim3(SCAN_BLOCKS), SCAN_THREADS, 0, stream>>>(
      xW_e, bWhh_e, 512, hS, cS, hB, enc_out, 1024, bar);
  cvt_kernel<<<g1((long)8192 * 1024), 256, 0, stream>>>(enc_out, 1024, enc_out_b, 1024,
                                                        8192, 1024);

  // decoder inputs: attn_in = [dec_out | ctx=hN+cN | dec_x]
  embed_dec_kernel<<<g1((long)2048 * 512), 256, 0, stream>>>(x_dec, emb, attn_in,
                                                             (long)2048 * 512);
  context_kernel<<<g1((long)2048 * 1024), 256, 0, stream>>>(hS, cS, attn_in,
                                                            (long)2048 * 1024);
  cvt_kernel<<<g1((long)2048 * 1536), 256, 0, stream>>>(attn_in + 1024, 3072, lstm_in,
                                                        1536, 2048, 1536);
  { int waves = (2048 / 32) * (4096 / 64);
    wmma_gemm_kernel<1><<<dim3((waves + 7) / 8), 256, 0, stream>>>(
        lstm_in, bWih_d, bsum_d, xW_d, 2048, 4096, 1536, 1536, 1536, 4096, 0, 0, 0); }
  lstm_scan_kernel<<<dim3(SCAN_BLOCKS), SCAN_THREADS, 0, stream>>>(
      xW_d, bWhh_d, 128, hS, cS, hB, attn_in, 3072, bar);  // h0=hN, c0=cN carry over

  // attention
  cvt_kernel<<<g1((long)2048 * 3072), 256, 0, stream>>>(attn_in, 3072, attn_b, 3072,
                                                        2048, 3072);
  { int waves = (2048 / 32) * (512 / 64);
    wmma_gemm_kernel<1><<<dim3((waves + 7) / 8), 256, 0, stream>>>(
        attn_b, bWa, ba, logits, 2048, 512, 3072, 3072, 3072, 512, 0, 0, 0); }
  softmax_kernel<<<dim3(2048), 256, 0, stream>>>(logits, softw, 512);
  { int waves = (128 / 32) * (1024 / 64);  // per-batch soft_w[b] @ enc_out[b]
    wmma_gemm_kernel<0><<<dim3((waves + 7) / 8, 1, 16), 256, 0, stream>>>(
        softw, enc_out_b, nullptr, htbar, 128, 1024, 512, 512, 1024, 1024,
        (long)128 * 512, (long)512 * 1024, (long)128 * 1024); }

  // gating + output projection
  cvt_kernel<<<g1((long)2048 * 1024), 256, 0, stream>>>(attn_in, 3072, catg, 2048,
                                                        2048, 1024);
  cvt_kernel<<<g1((long)2048 * 1024), 256, 0, stream>>>(htbar, 1024, catg + 1024, 2048,
                                                        2048, 1024);
  { int waves = (2048 / 32) * (1024 / 64);
    wmma_gemm_kernel<1><<<dim3((waves + 7) / 8), 256, 0, stream>>>(
        catg, bWal, bal, ailog, 2048, 1024, 2048, 2048, 2048, 1024, 0, 0, 0); }
  gate_kernel<<<g1((long)2048 * 1024), 256, 0, stream>>>(attn_in, htbar, ailog, outg,
                                                         (long)2048 * 1024);
  { int waves = (2048 / 32) * (32000 / 64);
    wmma_gemm_kernel<1><<<dim3((waves + 7) / 8), 256, 0, stream>>>(
        outg, bWo, bo, out_v, 2048, 32000, 1024, 1024, 1024, 32000, 0, 0, 0); }
  targets_kernel<<<g1(2048), 256, 0, stream>>>(tgt, 2048);
}
